// WindowEfficientSelfAttention_10668698763679
// MI455X (gfx1250) — compile-verified, matrix operand order checked by reading, not run on hardware
//
#include <hip/hip_runtime.h>
#include <hip/hip_bf16.h>
#include <math.h>

typedef __bf16 bf16_t;
typedef __attribute__((ext_vector_type(16))) __bf16 v16bf;
typedef __attribute__((ext_vector_type(8)))  __bf16 v8bf;
typedef __attribute__((ext_vector_type(8)))  float  v8f;

// Problem constants: B=2, C=64, H=W=512, WS=8, NH=16, QKC=32, ATTD=64, HID=256
#define CH_SCALE_F 0.70710678118654752f
#define INV_SQRT2_F 0.70710678118654752f

// ---------------------------------------------------------------------------
// WMMA helpers (wave32, 16x16x32 bf16 -> f32). Layouts per CDNA5 ISA 7.12.2.
// A 16x32: lane holds row M=lane%16; lanes<16 hold K {0..7,16..23},
//          lanes>=16 hold K {8..15,24..31}  -> two contiguous 8-elem runs.
// B fragments are pre-permuted to per-lane register order in the prep kernel.
// C/D f32: lane col N=lane%16; VGPR r holds row M = r + 8*(lane>=16).
// ---------------------------------------------------------------------------
__device__ __forceinline__ v16bf load_tileA(const bf16_t* base, int ld) {
  int lane = threadIdx.x & 31;
  int m = lane & 15, hi = lane >> 4;
  const bf16_t* row = base + m * ld + 8 * hi;
  v8bf c0 = *(const v8bf*)(row);        // 16B -> ds_load_b128
  v8bf c1 = *(const v8bf*)(row + 16);
  return __builtin_shufflevector(c0, c1, 0, 1, 2, 3, 4, 5, 6, 7,
                                 8, 9, 10, 11, 12, 13, 14, 15);
}

__device__ __forceinline__ v16bf load_fragB(const bf16_t* fragbuf, int tile) {
  int lane = threadIdx.x & 31;
  return *(const v16bf*)(fragbuf + (((size_t)tile * 32 + lane) << 4)); // 32B
}

__device__ __forceinline__ v8f wmma_bf16(v16bf a, v16bf b, v8f c) {
  return __builtin_amdgcn_wmma_f32_16x16x32_bf16(false, a, false, b, (short)0, c,
                                                 false, false);
}

// Async global->LDS copy, 16B per lane (tracked by ASYNCcnt). Generic pointers
// to __shared__ carry the LDS byte offset in their low 32 bits (flat-LDS
// aperture), which is exactly what VDST of the async op wants.
__device__ __forceinline__ void async_copy_b128(const void* gsrc, void* ldst) {
  unsigned lds_addr = (unsigned)(unsigned long long)ldst;
  asm volatile("global_load_async_to_lds_b128 %0, %1, off"
               :
               : "v"(lds_addr), "v"(gsrc)
               : "memory");
}

__device__ __forceinline__ void wait_asynccnt0() {
  asm volatile("s_wait_asynccnt 0x0" ::: "memory");
}

// k index inside a 32-wide K-slab for fragment element j on lane `lane`
__device__ __forceinline__ int frag_k(int lane, int j) {
  return ((j < 8) ? j : (j + 8)) + 8 * (lane >> 4);
}

// ---------------------------------------------------------------------------
// Kernel 0: fold BN into 1x1-conv weights and emit fragment-major bf16
// B-matrices (tile -> lane -> 16 contiguous elems) plus fused fp32 biases.
// Tile id = ks * Ntiles + nt.
// ---------------------------------------------------------------------------
__global__ __launch_bounds__(256) void prep_kernel(
    const float* bn1_g, const float* bn1_b, const float* bn1_m, const float* bn1_v,
    const float* q_w, const float* q_b, const float* k_w, const float* k_b,
    const float* v_w, const float* v_b, const float* hp_w,
    const float* bn2_g, const float* bn2_b, const float* bn2_m, const float* bn2_v,
    const float* fc1_w, const float* fc1_b,
    const float* bn3_g, const float* bn3_b, const float* bn3_m, const float* bn3_v,
    const float* fc2_w, const float* fc2_b,
    bf16_t* Wcf, float* biasC, bf16_t* Whpf,
    bf16_t* B1f, float* b1p, bf16_t* B2f, float* b2p)
{
  __shared__ float s1c[64], t1c[64], s2c[64], t2c[64], s3h[256], t3h[256];
  int tid = threadIdx.x;
  if (tid < 64) {
    float s = bn1_g[tid] * rsqrtf(bn1_v[tid] + 1e-5f);
    s1c[tid] = s; t1c[tid] = bn1_b[tid] - bn1_m[tid] * s;
    float s2 = bn2_g[tid] * rsqrtf(bn2_v[tid] + 1e-5f);
    s2c[tid] = s2; t2c[tid] = bn2_b[tid] - bn2_m[tid] * s2;
  }
  {
    float s3 = bn3_g[tid] * rsqrtf(bn3_v[tid] + 1e-5f);
    s3h[tid] = s3; t3h[tid] = bn3_b[tid] - bn3_m[tid] * s3;
  }
  __syncthreads();

  // QKV combined (K=64 ch, N=128 out; Ntiles=8, Ktiles=2) with bn1 folded
  for (int e = tid; e < 16 * 512; e += 256) {
    int tile = e >> 9, r = e & 511, lane = r >> 4, j = r & 15;
    int ks = tile >> 3, nt = tile & 7;
    int k = 32 * ks + frag_k(lane, j);
    int n = nt * 16 + (lane & 15);           // 0..31 q, 32..63 k, 64..127 v
    float w = (n < 32) ? q_w[n * 64 + k]
            : (n < 64) ? k_w[(n - 32) * 64 + k]
                       : v_w[(n - 64) * 64 + k];
    Wcf[e] = (bf16_t)(w * s1c[k]);
  }
  if (tid < 128) {
    int o = tid;
    float acc = (o < 32) ? q_b[o] : (o < 64) ? k_b[o - 32] : v_b[o - 64];
    for (int c = 0; c < 64; ++c) {
      float w = (o < 32) ? q_w[o * 64 + c]
              : (o < 64) ? k_w[(o - 32) * 64 + c]
                         : v_w[(o - 64) * 64 + c];
      acc += w * t1c[c];
    }
    biasC[o] = acc;
  }
  // hp (K=64, N=64; Ntiles=4, Ktiles=2)
  for (int e = tid; e < 8 * 512; e += 256) {
    int tile = e >> 9, r = e & 511, lane = r >> 4, j = r & 15;
    int ks = tile >> 2, nt = tile & 3;
    int k = 32 * ks + frag_k(lane, j);
    int n = nt * 16 + (lane & 15);
    Whpf[e] = (bf16_t)hp_w[n * 64 + k];
  }
  // fc1 with bn2 folded (K=64, N=256; Ntiles=16, Ktiles=2)
  for (int e = tid; e < 32 * 512; e += 256) {
    int tile = e >> 9, r = e & 511, lane = r >> 4, j = r & 15;
    int ks = tile >> 4, nt = tile & 15;
    int k = 32 * ks + frag_k(lane, j);
    int n = nt * 16 + (lane & 15);
    B1f[e] = (bf16_t)(fc1_w[n * 64 + k] * s2c[k]);
  }
  {
    int h = tid;
    float acc = fc1_b[h];
    for (int c = 0; c < 64; ++c) acc += fc1_w[h * 64 + c] * t2c[c];
    b1p[h] = acc;
  }
  // fc2 with bn3 folded (K=256, N=64; Ntiles=4, Ktiles=8)
  for (int e = tid; e < 32 * 512; e += 256) {
    int tile = e >> 9, r = e & 511, lane = r >> 4, j = r & 15;
    int ks = tile >> 2, nt = tile & 3;
    int k = 32 * ks + frag_k(lane, j);
    int n = nt * 16 + (lane & 15);
    B2f[e] = (bf16_t)(fc2_w[n * 256 + k] * s3h[k]);
  }
  if (tid < 64) {
    int o = tid;
    float acc = fc2_b[o];
    for (int h = 0; h < 256; ++h) acc += fc2_w[o * 256 + h] * t3h[h];
    b2p[o] = acc;
  }
}

// ---------------------------------------------------------------------------
// Kernel 1: PA gate, window-blocked. One 8x8 window per block. Emits:
//   shortcut (f32, NCHW)             - for the attn residual add
//   xwin     (bf16, [win][n=64][c])  - contiguous 8KB/window, pixel-major:
//                                      exactly the WMMA A layout attn needs.
// ---------------------------------------------------------------------------
__global__ __launch_bounds__(256) void pa_kernel(const float* __restrict__ x,
                                                 const float* __restrict__ pa_w,
                                                 const float* __restrict__ pa_b,
                                                 float* __restrict__ shortcut,
                                                 bf16_t* __restrict__ xwin)
{
  __shared__ __align__(16) bf16_t xt[64][72];

  int tid = threadIdx.x;
  int w = blockIdx.x;
  int b = w >> 12;
  int rr = w & 4095;
  int wy = rr >> 6, wx = rr & 63;
  int row0 = wy * 8, col0 = wx * 8;

  int c = tid >> 2;           // channel
  int q = tid & 3;            // quarter: pixels q*16 .. q*16+15
  const float* xp = x + (((size_t)(b * 64 + c)) << 18);
  float* op = shortcut + (((size_t)(b * 64 + c)) << 18);
  float wk[9];
#pragma unroll
  for (int i = 0; i < 9; ++i) wk[i] = pa_w[c * 9 + i];
  float bias = pa_b[c];

#pragma unroll
  for (int i = 0; i < 16; ++i) {
    int n = q * 16 + i;
    int row = row0 + (n >> 3), col = col0 + (n & 7);
    float acc = bias;
#pragma unroll
    for (int ky = 0; ky < 3; ++ky) {
      int y = row + ky - 1;
      if (y < 0 || y > 511) continue;
#pragma unroll
      for (int kx = 0; kx < 3; ++kx) {
        int xx = col + kx - 1;
        if (xx < 0 || xx > 511) continue;
        acc += wk[ky * 3 + kx] * xp[(y << 9) + xx];
      }
    }
    float xv = xp[(row << 9) + col];
    float g = xv * (1.0f / (1.0f + __expf(-acc)));
    op[(row << 9) + col] = g;
    xt[n][c] = (bf16_t)g;
  }
  __syncthreads();

  // LDS -> global, contiguous 16B chunks (coalesced)
  for (int s = tid; s < 512; s += 256) {
    int n = s >> 3, c8 = s & 7;
    v8bf val = *(const v8bf*)&xt[n][c8 * 8];
    *(v8bf*)(xwin + (((size_t)w) << 12) + (size_t)s * 8) = val;
  }
}

// ---------------------------------------------------------------------------
// Kernel 2: windowed efficient attention; one 8x8 window per 256-thread block.
// Window tile arrives via async global->LDS b128 copies (ASYNCcnt path).
// ---------------------------------------------------------------------------
__global__ __launch_bounds__(256) void attn_kernel(
    const float* __restrict__ shortcut, const bf16_t* __restrict__ xwin,
    const bf16_t* __restrict__ Wcf, const float* __restrict__ biasC,
    const bf16_t* __restrict__ Whpf, const float* __restrict__ hp_b,
    float* __restrict__ s1)
{
  __shared__ __align__(16) bf16_t xw[64][72];     // pixels(n) x channels, padded
  __shared__ __align__(16) float qkv[64][132];    // q:0..31 k:32..63 v:64..127
  __shared__ float ctxs[16][2][4];
  __shared__ float dots_s[16][4];
  __shared__ float dcs[16];
  __shared__ __align__(16) bf16_t attn_s[64][72];

  int tid = threadIdx.x;
  int wid = tid >> 5;
  int lane = tid & 31;
  int nlane = lane & 15, hi = lane >> 4;

  int w = blockIdx.x;
  int b = w >> 12;
  int rr = w & 4095;
  int wy = rr >> 6, wx = rr & 63;
  int row0 = wy * 8, col0 = wx * 8;

  // Async-copy the 8KB window into padded LDS: 512 x 16B segments, per-lane
  // LDS destinations keep the +8 padding.
#pragma unroll
  for (int it = 0; it < 2; ++it) {
    int s = it * 256 + tid;
    int n = s >> 3, c8 = s & 7;
    async_copy_b128(xwin + (((size_t)w) << 12) + (size_t)s * 8, &xw[n][c8 * 8]);
  }
  wait_asynccnt0();
  __syncthreads();

  // QKV GEMM: 4 M-tiles x 8 N-tiles, K=64 (2 steps). Wave w owns N-tile w.
  {
    v16bf b0 = load_fragB(Wcf, 0 * 8 + wid);
    v16bf b1 = load_fragB(Wcf, 1 * 8 + wid);
    int o = wid * 16 + nlane;
    float bsv = biasC[o];
#pragma unroll
    for (int mt = 0; mt < 4; ++mt) {
      v16bf a0 = load_tileA(&xw[mt * 16][0], 72);
      v16bf a1 = load_tileA(&xw[mt * 16][32], 72);
      v8f acc = {};
      acc = wmma_bf16(a0, b0, acc);
      acc = wmma_bf16(a1, b1, acc);
#pragma unroll
      for (int r = 0; r < 8; ++r)
        qkv[mt * 16 + r + 8 * hi][o] = acc[r] + bsv;
    }
  }
  __syncthreads();

  // q softmax over dk=2 per (n, head)
  for (int p = tid; p < 1024; p += 256) {
    int n = p >> 4, h = p & 15;
    float a = qkv[n][2 * h], bq = qkv[n][2 * h + 1];
    float m = fmaxf(a, bq);
    float ea = __expf(a - m), eb = __expf(bq - m);
    float inv = 1.0f / (ea + eb);
    qkv[n][2 * h] = ea * inv;
    qkv[n][2 * h + 1] = eb * inv;
  }
  // k softmax over n=64 per column
  if (tid < 32) {
    int o = 32 + tid;
    float m = -1e30f;
    for (int n = 0; n < 64; ++n) m = fmaxf(m, qkv[n][o]);
    float s = 0.f;
    for (int n = 0; n < 64; ++n) {
      float e = __expf(qkv[n][o] - m);
      qkv[n][o] = e; s += e;
    }
    float inv = 1.0f / s;
    for (int n = 0; n < 64; ++n) qkv[n][o] *= inv;
  }
  __syncthreads();

  // ctx[h][d][e] = sum_n k*v ; dots[h][d][d2] = sum_n q*k
  if (tid < 192) {
    if (tid < 128) {
      int h = tid >> 3, d = (tid >> 2) & 1, e = tid & 3;
      float acc = 0.f;
      for (int n = 0; n < 64; ++n)
        acc += qkv[n][32 + 2 * h + d] * qkv[n][64 + 4 * h + e];
      ctxs[h][d][e] = acc;
    } else {
      int t2 = tid - 128;
      int h = t2 >> 2, d = (t2 >> 1) & 1, d2 = t2 & 1;
      float acc = 0.f;
      for (int n = 0; n < 64; ++n)
        acc += qkv[n][2 * h + d] * qkv[n][32 + 2 * h + d2];
      dots_s[h][2 * d + d2] = acc;
    }
  }
  __syncthreads();
  // dc[h] = max + mean over the 2x2 (transpose irrelevant for max/mean)
  if (tid < 16) {
    float v0 = dots_s[tid][0] * CH_SCALE_F;
    float v1 = dots_s[tid][1] * CH_SCALE_F;
    float v2 = dots_s[tid][2] * CH_SCALE_F;
    float v3 = dots_s[tid][3] * CH_SCALE_F;
    float mx = fmaxf(fmaxf(v0, v1), fmaxf(v2, v3));
    dcs[tid] = mx + 0.25f * (v0 + v1 + v2 + v3);
  }
  __syncthreads();

  // attn[n][o] = dc[h] * (q @ ctx), o = 4h + e
  for (int p = tid; p < 4096; p += 256) {
    int n = p >> 6, o = p & 63;
    int h = o >> 2, e = o & 3;
    float val = dcs[h] * (qkv[n][2 * h] * ctxs[h][0][e] +
                          qkv[n][2 * h + 1] * ctxs[h][1][e]);
    attn_s[n][o] = (bf16_t)val;
  }
  __syncthreads();

  // hp projection + residual: 4 M x 4 N tiles. Wave -> (nt = wid&3, 2 M-tiles)
  {
    int nt = wid & 3;
    v16bf b0 = load_fragB(Whpf, 0 * 4 + nt);
    v16bf b1 = load_fragB(Whpf, 1 * 4 + nt);
    int o = nt * 16 + nlane;
    float bsv = hp_b[o];
#pragma unroll
    for (int i = 0; i < 2; ++i) {
      int mt = (wid >> 2) * 2 + i;
      v16bf a0 = load_tileA(&attn_s[mt * 16][0], 72);
      v16bf a1 = load_tileA(&attn_s[mt * 16][32], 72);
      v8f acc = {};
      acc = wmma_bf16(a0, b0, acc);
      acc = wmma_bf16(a1, b1, acc);
#pragma unroll
      for (int r = 0; r < 8; ++r) {
        int npx = mt * 16 + r + 8 * hi;
        int row = row0 + (npx >> 3), col = col0 + (npx & 7);
        size_t g = (((size_t)(b * 64 + o)) << 18) + (row << 9) + col;
        s1[g] = acc[r] + bsv + shortcut[g];
      }
    }
  }
}

// ---------------------------------------------------------------------------
// Kernel 3: MLP fused per 16x8 tile with 1-px halo (18x10=180 px, padded 192).
// Dynamic LDS: a_s1[192][72] + hid[192][264] + ybuf[128][264] bf16 = 192 KB.
// ---------------------------------------------------------------------------
__global__ __launch_bounds__(256) void mlp_kernel(
    const float* __restrict__ s1, const bf16_t* __restrict__ B1f,
    const float* __restrict__ b1p, const float* __restrict__ dw_w,
    const float* __restrict__ dw_b, const bf16_t* __restrict__ B2f,
    const float* __restrict__ b2p, float* __restrict__ out)
{
  extern __shared__ __align__(16) char smem[];
  bf16_t* a_s1 = (bf16_t*)smem;              // [192][72]
  bf16_t* hid  = a_s1 + 192 * 72;            // [192][264]
  bf16_t* ybuf = hid + 192 * 264;            // [128][264]

  int tid = threadIdx.x;
  int wid = tid >> 5;
  int lane = tid & 31;
  int nlane = lane & 15, hi = lane >> 4;

  int blk = blockIdx.x;
  int b = blk >> 11;
  int rr = blk & 2047;
  int ty = rr >> 5, tx = rr & 31;
  int y0 = ty * 8, x0 = tx * 16;

  // Halo load: lanes -> consecutive hx (coalesced); rows 180..191 zero
  for (int e = tid; e < 192 * 64; e += 256) {
    int c = e / 192, p = e - c * 192;
    float v = 0.f;
    if (p < 180) {
      int hy = p / 18, hx = p - hy * 18;
      int gy = y0 + hy - 1, gx = x0 + hx - 1;
      if (gy >= 0 && gy < 512 && gx >= 0 && gx < 512)
        v = s1[(((size_t)(b * 64 + c)) << 18) + (gy << 9) + gx];
    }
    a_s1[p * 72 + c] = (bf16_t)v;
  }
  __syncthreads();

  // fc1 (bn2 folded) + exact GELU: 12 M x 16 N tiles, K=64.
  // Wave w owns N-tiles w and w+8; B-frags held in registers across M.
#pragma unroll
  for (int half = 0; half < 2; ++half) {
    int nt = wid + half * 8;
    v16bf b0 = load_fragB(B1f, 0 * 16 + nt);
    v16bf b1 = load_fragB(B1f, 1 * 16 + nt);
    int o = nt * 16 + nlane;
    float bsv = b1p[o];
#pragma unroll
    for (int mt = 0; mt < 12; ++mt) {
      v16bf a0 = load_tileA(a_s1 + (mt * 16) * 72, 72);
      v16bf a1 = load_tileA(a_s1 + (mt * 16) * 72 + 32, 72);
      v8f acc = {};
      acc = wmma_bf16(a0, b0, acc);
      acc = wmma_bf16(a1, b1, acc);
#pragma unroll
      for (int r = 0; r < 8; ++r) {
        int m = mt * 16 + r + 8 * hi;
        float xg = acc[r] + bsv;
        float g = 0.5f * xg * (1.0f + erff(xg * INV_SQRT2_F));
        hid[m * 264 + o] = (bf16_t)g;
      }
    }
  }
  __syncthreads();

  // Depthwise 3x3 + bias (bn3 folded into fc2 weights)
  for (int e = tid; e < 128 * 256; e += 256) {
    int px = e >> 8, c = e & 255;
    int tyi = px >> 4, txi = px & 15;
    float acc = dw_b[c];
#pragma unroll
    for (int ky = 0; ky < 3; ++ky)
#pragma unroll
      for (int kx = 0; kx < 3; ++kx) {
        int p = (tyi + ky) * 18 + (txi + kx);
        acc += dw_w[c * 9 + ky * 3 + kx] * (float)hid[p * 264 + c];
      }
    ybuf[px * 264 + c] = (bf16_t)acc;
  }
  __syncthreads();

  // fc2 + residual: 8 M x 4 N tiles, K=256 (8 steps). Wave w owns M-tile w;
  // its 8 A-fragments stay in registers across the N loop.
  {
    int mt = wid;
    v16bf af[8];
#pragma unroll
    for (int ks = 0; ks < 8; ++ks)
      af[ks] = load_tileA(ybuf + (mt * 16) * 264 + ks * 32, 264);
#pragma unroll
    for (int nt = 0; nt < 4; ++nt) {
      v8f acc = {};
#pragma unroll
      for (int ks = 0; ks < 8; ++ks)
        acc = wmma_bf16(af[ks], load_fragB(B2f, ks * 4 + nt), acc);
      int o = nt * 16 + nlane;
      float bsv = b2p[o];
#pragma unroll
      for (int r = 0; r < 8; ++r) {
        int px = mt * 16 + r + 8 * hi;
        int gy = y0 + (px >> 4), gx = x0 + (px & 15);
        size_t g = (((size_t)(b * 64 + o)) << 18) + (gy << 9) + gx;
        out[g] = acc[r] + bsv + s1[g];
      }
    }
  }
}

// ---------------------------------------------------------------------------
extern "C" void kernel_launch(void* const* d_in, const int* in_sizes, int n_in,
                              void* d_out, int out_size, void* d_ws, size_t ws_size,
                              hipStream_t stream) {
  (void)in_sizes; (void)n_in; (void)out_size; (void)ws_size;
  const float* x     = (const float*)d_in[0];
  const float* pa_w  = (const float*)d_in[1];
  const float* pa_b  = (const float*)d_in[2];
  const float* bn1_g = (const float*)d_in[3];
  const float* bn1_b = (const float*)d_in[4];
  const float* bn1_m = (const float*)d_in[5];
  const float* bn1_v = (const float*)d_in[6];
  const float* q_w   = (const float*)d_in[7];
  const float* q_b   = (const float*)d_in[8];
  const float* k_w   = (const float*)d_in[9];
  const float* k_b   = (const float*)d_in[10];
  const float* v_w   = (const float*)d_in[11];
  const float* v_b   = (const float*)d_in[12];
  const float* hp_w  = (const float*)d_in[13];
  const float* hp_b  = (const float*)d_in[14];
  const float* bn2_g = (const float*)d_in[15];
  const float* bn2_b = (const float*)d_in[16];
  const float* bn2_m = (const float*)d_in[17];
  const float* bn2_v = (const float*)d_in[18];
  const float* fc1_w = (const float*)d_in[19];
  const float* fc1_b = (const float*)d_in[20];
  const float* dw_w  = (const float*)d_in[21];
  const float* dw_b  = (const float*)d_in[22];
  const float* bn3_g = (const float*)d_in[23];
  const float* bn3_b = (const float*)d_in[24];
  const float* bn3_m = (const float*)d_in[25];
  const float* bn3_v = (const float*)d_in[26];
  const float* fc2_w = (const float*)d_in[27];
  const float* fc2_b = (const float*)d_in[28];

  char* ws = (char*)d_ws;
  size_t off = 0;
  auto take = [&](size_t bytes) -> char* {
    char* p = ws + off;
    off = (off + bytes + 255) & ~(size_t)255;
    return p;
  };
  float*  shortcut = (float*) take((size_t)2 * 64 * 512 * 512 * 4);  // 134 MB
  float*  s1buf    = (float*) take((size_t)2 * 64 * 512 * 512 * 4);  // 134 MB
  bf16_t* xwin     = (bf16_t*)take((size_t)8192 * 4096 * 2);         //  67 MB
  bf16_t* Wcf      = (bf16_t*)take(16 * 512 * 2);
  float*  biasC    = (float*) take(128 * 4);
  bf16_t* Whpf     = (bf16_t*)take(8 * 512 * 2);
  bf16_t* B1f      = (bf16_t*)take(32 * 512 * 2);
  float*  b1p      = (float*) take(256 * 4);
  bf16_t* B2f      = (bf16_t*)take(32 * 512 * 2);
  float*  b2p      = (float*) take(64 * 4);

  prep_kernel<<<1, 256, 0, stream>>>(
      bn1_g, bn1_b, bn1_m, bn1_v, q_w, q_b, k_w, k_b, v_w, v_b, hp_w,
      bn2_g, bn2_b, bn2_m, bn2_v, fc1_w, fc1_b,
      bn3_g, bn3_b, bn3_m, bn3_v, fc2_w, fc2_b,
      Wcf, biasC, Whpf, B1f, b1p, B2f, b2p);

  pa_kernel<<<8192, 256, 0, stream>>>(x, pa_w, pa_b, shortcut, xwin);

  attn_kernel<<<8192, 256, 0, stream>>>(shortcut, xwin, Wcf, biasC, Whpf, hp_b,
                                        s1buf);

  size_t smem = (size_t)(192 * 72 + 192 * 264 + 128 * 264) * sizeof(bf16_t);
  mlp_kernel<<<2 * 64 * 32, 256, smem, stream>>>(s1buf, B1f, b1p, dw_w, dw_b,
                                                 B2f, b2p, (float*)d_out);
}
